// PerformerViTTactile_3977139716168
// MI455X (gfx1250) — matrix operand order repaired and checked by reading, not compiled, not run on hardware
//
#include <hip/hip_runtime.h>
#include <math.h>

// ---------------------------------------------------------------------------
// ViT-Tiny + Performer attention forward for MI455X (gfx1250, wave32).
// All GEMMs run through v_wmma_f32_16x16x32_bf16 (bf16 A/B, f32 accum).
// Block tile 64x64, 8 waves; each wave computes two 16x16 subtiles reusing
// its A fragment (2 WMMA per K-step). Fast paths: global_load_b128 +
// ds_store_b128; fragments read as contiguous ds_load_b128.
// ---------------------------------------------------------------------------

typedef unsigned short u16;
typedef __attribute__((ext_vector_type(16))) __bf16 v16bf;
typedef __attribute__((ext_vector_type(8)))  float  v8f;

#define DEVI static __device__ __forceinline__

DEVI u16 f2bf(float f) {            // round-to-nearest-even f32 -> bf16 bits
  unsigned int u = __builtin_bit_cast(unsigned int, f);
  u += 0x7fffu + ((u >> 16) & 1u);
  return (u16)(u >> 16);
}
DEVI float bf2f(u16 h) {
  unsigned int u = ((unsigned int)h) << 16;
  return __builtin_bit_cast(float, u);
}

enum { EPI_NONE = 0, EPI_GELU = 1, EPI_EXP = 2, EPI_ROWSCALE = 3, EPI_ACCUM = 4 };

// ---------------------------------------------------------------------------
// Generic batched bf16 WMMA GEMM:  C[M,N] (+)= epi( A[M,K] * B[K,N] + bias )
// A row-major (lda); B row-major (ldb) or, if TRANSB, B given as [N,K] (ldb).
// Batch z decomposes as z1 = z/batch2, z2 = z%batch2, independent strides.
// Alignment contract: lda % 8 == 0, ldb % 8 == 0 (TRANSB) / % 8 (else),
// base pointers 256B aligned, batch strides % 8 == 0.
// ---------------------------------------------------------------------------
template <bool TRANSB>
__global__ __launch_bounds__(256) void wmma_gemm(
    const u16* __restrict__ A, long long sA1, long long sA2, int lda,
    const u16* __restrict__ Bm, long long sB1, long long sB2, int ldb,
    float* C, long long sC1, long long sC2, int ldc,
    u16* Cb, long long sCb1, long long sCb2, int ldcb,
    const float* __restrict__ bias,
    const float* __restrict__ rowscale, long long sR1, long long sR2,
    int M, int N, int K, int batch2, int epi, float scale) {
  __shared__ u16 As[64][40];    // row stride 80B (16B multiple)
  __shared__ u16 BsT[64][40];   // stored transposed: BsT[n][k]

  const int t = threadIdx.x;
  const int z = blockIdx.z;
  const int z1 = z / batch2, z2 = z - z1 * batch2;
  A  += (long long)z1 * sA1 + (long long)z2 * sA2;
  Bm += (long long)z1 * sB1 + (long long)z2 * sB2;
  if (C)        C        += (long long)z1 * sC1  + (long long)z2 * sC2;
  if (Cb)       Cb       += (long long)z1 * sCb1 + (long long)z2 * sCb2;
  if (rowscale) rowscale += (long long)z1 * sR1  + (long long)z2 * sR2;

  const int bm0 = blockIdx.y * 64;
  const int bn0 = blockIdx.x * 64;
  const int wave = t >> 5, lane = t & 31;
  const int wm = wave & 3, wn = wave >> 2;       // 4 M-tiles x 2 N-groups (32 cols each)
  const int half = lane >> 4, r = lane & 15;

  // staging roles (both tiles are 64 x 32 = 2048 u16, 8 per thread)
  const int ar = t >> 2, ac = (t & 3) * 8;       // A: row ar, k ac..ac+7
  const int bkk = t >> 3, bn8 = (t & 7) * 8;     // B row-major: k bkk, n bn8..+7

  const bool fullM = (bm0 + 64 <= M);
  const bool fullN = (bn0 + 64 <= N);

  v8f acc0 = {}, acc1 = {};
  union Frag { v16bf v; uint4 q[2]; } fa, fb0, fb1;

  for (int k0 = 0; k0 < K; k0 += 32) {
    const bool fullK = (k0 + 32 <= K);

    { // ---- stage A tile (64 rows x 32 k) ----
      const int gr = bm0 + ar;
      const long long rowb = (long long)gr * lda + k0 + ac;
      if (fullM && fullK) {                      // block-uniform fast path
        *(uint4*)&As[ar][ac] = *(const uint4*)(A + rowb);
      } else {
#pragma unroll
        for (int j = 0; j < 8; ++j) {
          int gc = k0 + ac + j;
          As[ar][ac + j] = (gr < M && gc < K) ? A[rowb + j] : (u16)0;
        }
      }
    }
    if (TRANSB) { // ---- B as [N,K]: contiguous in K, store straight ----
      const int n = t >> 2, kk = (t & 3) * 8;    // 64 rows x 32 k
      const int gn = bn0 + n;
      const long long rowb = (long long)gn * ldb + k0 + kk;
      if (fullN && fullK) {
        *(uint4*)&BsT[n][kk] = *(const uint4*)(Bm + rowb);
      } else {
#pragma unroll
        for (int j = 0; j < 8; ++j) {
          int gk = k0 + kk + j;
          BsT[n][kk + j] = (gk < K && gn < N) ? Bm[rowb + j] : (u16)0;
        }
      }
    } else {      // ---- B row-major [K,N]: coalesced 16B load, scatter-transpose ----
      const int gk = k0 + bkk;
      const long long rowb = (long long)gk * ldb + bn0 + bn8;
      if (fullN && fullK) {
        uint4 v = *(const uint4*)(Bm + rowb);
        const u16* pv = (const u16*)&v;
#pragma unroll
        for (int j = 0; j < 8; ++j) BsT[bn8 + j][bkk] = pv[j];
      } else {
#pragma unroll
        for (int j = 0; j < 8; ++j) {
          int gn = bn0 + bn8 + j;
          BsT[bn8 + j][bkk] = (gk < K && gn < N) ? Bm[rowb + j] : (u16)0;
        }
      }
    }
    __syncthreads();

    // Fragments in contiguous k order (k = 16*half + s), same for A and B — all reads contiguous & 16B aligned.
    const int arow = wm * 16 + r;
    fa.q[0] = *(const uint4*)&As[arow][half * 16];       // K 0..7  / 16..23
    fa.q[1] = *(const uint4*)&As[arow][half * 16 + 8];   // K 8..15 / 24..31
    const int bc0 = wn * 32 + r, bc1 = bc0 + 16;
    fb0.q[0] = *(const uint4*)&BsT[bc0][half * 16];
    fb0.q[1] = *(const uint4*)&BsT[bc0][half * 16 + 8];
    fb1.q[0] = *(const uint4*)&BsT[bc1][half * 16];
    fb1.q[1] = *(const uint4*)&BsT[bc1][half * 16 + 8];

    acc0 = __builtin_amdgcn_wmma_f32_16x16x32_bf16(
        false, fa.v, false, fb0.v, (short)0, acc0, false, false);
    acc1 = __builtin_amdgcn_wmma_f32_16x16x32_bf16(
        false, fa.v, false, fb1.v, (short)0, acc1, false, false);
    __syncthreads();
  }

  // Epilogue: D layout — lane half picks row block, VGPR j = row offset.
  const int col0 = bn0 + wn * 32 + r;
  const int col1 = col0 + 16;
#pragma unroll
  for (int j = 0; j < 8; ++j) {
    int row = bm0 + wm * 16 + half * 8 + j;
    if (row >= M) continue;
    float rsv = rowscale ? rowscale[row] : 0.f;
    float vv[2] = {acc0[j], acc1[j]};
    int cc[2] = {col0, col1};
#pragma unroll
    for (int p = 0; p < 2; ++p) {
      int col = cc[p];
      if (col >= N) continue;
      float v = vv[p];
      if (bias) v += bias[col];
      if (epi == EPI_GELU)          v = 0.5f * v * (1.0f + erff(v * 0.7071067811865475f));
      else if (epi == EPI_EXP)      v = __expf(v - rsv) * scale;
      else if (epi == EPI_ROWSCALE) v *= rsv;
      if (C) {
        long long ci = (long long)row * ldc + col;
        if (epi == EPI_ACCUM) C[ci] += v; else C[ci] = v;
      }
      if (Cb) Cb[(long long)row * ldcb + col] = f2bf(v);
    }
  }
}

// ---------------------------------------------------------------------------
// Helper kernels
// ---------------------------------------------------------------------------
__global__ void k_f32_to_bf16(const float* __restrict__ in, u16* __restrict__ out,
                              long long n) {
  long long i = (long long)blockIdx.x * blockDim.x + threadIdx.x;
  if (i < n) out[i] = f2bf(in[i]);
}

// patch_w [192,768] -> bf16 transposed [768,192]
__global__ void k_patchw_T(const float* __restrict__ in, u16* __restrict__ out) {
  int i = blockIdx.x * blockDim.x + threadIdx.x;
  if (i >= 768 * 192) return;
  int kk = i / 192, e = i % 192;
  out[i] = f2bf(in[e * 768 + kk]);
}

// im2col for non-overlapping 16x16 patches: A_pe[b*196+p][c*256+py*16+px]
__global__ void k_im2col(const float* __restrict__ x, u16* __restrict__ ap) {
  long long i = (long long)blockIdx.x * blockDim.x + threadIdx.x;
  if (i >= (long long)12544 * 768) return;
  int rowp = (int)(i / 768), cc = (int)(i % 768);
  int b = rowp / 196, p = rowp % 196;
  int gy = p / 14, gx = p % 14;
  int c = cc / 256, rem = cc % 256, py = rem / 16, px = rem % 16;
  float v = x[(((long long)b * 3 + c) * 224 + gy * 16 + py) * 224 + gx * 16 + px];
  ap[i] = f2bf(v);
}

// h[b,0,:]=cls+pos0 ; h[b,1+p,:]=LN(tok)+pos.  One wave per token row.
__global__ void k_assemble(const float* __restrict__ tok, const float* __restrict__ cls,
                           const float* __restrict__ pos, const float* __restrict__ pw,
                           const float* __restrict__ pb, float* __restrict__ h) {
  int wid = threadIdx.x >> 5, lane = threadIdx.x & 31;
  int row = blockIdx.x * 8 + wid;
  if (row >= 64 * 197) return;
  int n = row % 197;
  float* hr = h + (long long)row * 192;
  if (n == 0) {
#pragma unroll
    for (int j = 0; j < 6; ++j) { int c = lane + j * 32; hr[c] = cls[c] + pos[c]; }
    return;
  }
  int b = row / 197;
  const float* tr = tok + ((long long)b * 196 + (n - 1)) * 192;
  float vals[6]; float s = 0.f;
#pragma unroll
  for (int j = 0; j < 6; ++j) { vals[j] = tr[lane + j * 32]; s += vals[j]; }
  for (int o = 16; o; o >>= 1) s += __shfl_xor(s, o, 32);
  float mu = s * (1.f / 192.f);
  float vs = 0.f;
#pragma unroll
  for (int j = 0; j < 6; ++j) { float d = vals[j] - mu; vs += d * d; }
  for (int o = 16; o; o >>= 1) vs += __shfl_xor(vs, o, 32);
  float inv = rsqrtf(vs * (1.f / 192.f) + 1e-5f);
#pragma unroll
  for (int j = 0; j < 6; ++j) {
    int c = lane + j * 32;
    hr[c] = (vals[j] - mu) * inv * pw[c] + pb[c] + pos[(long long)n * 192 + c];
  }
}

// LayerNorm over width 192, f32 in -> bf16 out. One wave per row.
__global__ void k_ln(const float* __restrict__ x, const float* __restrict__ w,
                     const float* __restrict__ b, u16* __restrict__ out, int rows) {
  int wid = threadIdx.x >> 5, lane = threadIdx.x & 31;
  int row = blockIdx.x * 8 + wid;
  if (row >= rows) return;
  const float* xr = x + (long long)row * 192;
  float vals[6]; float s = 0.f;
#pragma unroll
  for (int j = 0; j < 6; ++j) { vals[j] = xr[lane + j * 32]; s += vals[j]; }
  for (int o = 16; o; o >>= 1) s += __shfl_xor(s, o, 32);
  float mu = s * (1.f / 192.f);
  float vs = 0.f;
#pragma unroll
  for (int j = 0; j < 6; ++j) { float d = vals[j] - mu; vs += d * d; }
  for (int o = 16; o; o >>= 1) vs += __shfl_xor(vs, o, 32);
  float inv = rsqrtf(vs * (1.f / 192.f) + 1e-5f);
#pragma unroll
  for (int j = 0; j < 6; ++j) {
    int c = lane + j * 32;
    out[(long long)row * 192 + c] = f2bf((vals[j] - mu) * inv * w[c] + b[c]);
  }
}

// 0.5*||q||^2 and 0.5*||k||^2 per (b,h,n), read from bf16 qkv buffer.
__global__ void k_rownorm(const u16* __restrict__ qkv, float* __restrict__ qn,
                          float* __restrict__ kn) {
  int i = blockIdx.x * blockDim.x + threadIdx.x;
  if (i >= 37824) return;              // 64*3*197, layout (b*3+h)*197+n
  int b = i / 591, rm = i % 591, hh = rm / 197, n = rm % 197;
  const u16* base = qkv + ((long long)(b * 197 + n)) * 576 + hh * 64;
  float sq = 0.f, sk = 0.f;
  for (int d = 0; d < 64; ++d) {
    float q = bf2f(base[d]);       sq += q * q;
    float k = bf2f(base[192 + d]); sk += k * k;
  }
  qn[i] = 0.5f * sq;
  kn[i] = 0.5f * sk;
}

// k_cumsum: kc[(b*3+h)*256+f] = sum_n pk[b,h,n,f]
__global__ void k_kcsum(const u16* __restrict__ pk, float* __restrict__ kc) {
  int i = blockIdx.x * blockDim.x + threadIdx.x;
  if (i >= 64 * 3 * 256) return;
  int bh = i >> 8, f = i & 255;
  const u16* base = pk + (long long)bh * 197 * 256 + f;
  float s = 0.f;
  for (int n = 0; n < 197; ++n) s += bf2f(base[n * 256]);
  kc[i] = s;
}

// D_inv[(b*3+h)*197+n] = 1 / dot(pq_row, kc_row)
__global__ void k_dinv(const u16* __restrict__ pq, const float* __restrict__ kc,
                       float* __restrict__ di) {
  int i = blockIdx.x * blockDim.x + threadIdx.x;
  if (i >= 37824) return;
  int bh = i / 197;
  const u16* row = pq + (long long)i * 256;
  const float* kcr = kc + (long long)bh * 256;
  float s = 0.f;
  for (int f = 0; f < 256; ++f) s += bf2f(row[f]) * kcr[f];
  di[i] = 1.0f / s;
}

// token mean-pool: pooled_bf[b,c] = mean_n h[b,n,c]
__global__ void k_pool(const float* __restrict__ h, u16* __restrict__ pooled) {
  int b = blockIdx.x, c = threadIdx.x;   // blockDim = 192
  const float* base = h + (long long)b * 197 * 192 + c;
  float s = 0.f;
  for (int n = 0; n < 197; ++n) s += base[n * 192];
  pooled[b * 192 + c] = f2bf(s * (1.0f / 197.0f));
}

// ---------------------------------------------------------------------------
// Host-side GEMM dispatcher
// ---------------------------------------------------------------------------
static void launch_gemm(hipStream_t s, bool transb, int epi,
                        const u16* A, long long sA1, long long sA2, int lda,
                        const u16* Bm, long long sB1, long long sB2, int ldb,
                        float* C, long long sC1, long long sC2, int ldc,
                        u16* Cb, long long sCb1, long long sCb2, int ldcb,
                        const float* bias, const float* rowscale,
                        long long sR1, long long sR2,
                        int M, int N, int K, int b1, int b2, float scale) {
  dim3 g((unsigned)((N + 63) / 64), (unsigned)((M + 63) / 64), (unsigned)(b1 * b2));
  dim3 t(256, 1, 1);
  if (transb)
    wmma_gemm<true><<<g, t, 0, s>>>(A, sA1, sA2, lda, Bm, sB1, sB2, ldb,
                                    C, sC1, sC2, ldc, Cb, sCb1, sCb2, ldcb,
                                    bias, rowscale, sR1, sR2, M, N, K, b2, epi, scale);
  else
    wmma_gemm<false><<<g, t, 0, s>>>(A, sA1, sA2, lda, Bm, sB1, sB2, ldb,
                                     C, sC1, sC2, ldc, Cb, sCb1, sCb2, ldcb,
                                     bias, rowscale, sR1, sR2, M, N, K, b2, epi, scale);
}

// ---------------------------------------------------------------------------
// Orchestration
// ---------------------------------------------------------------------------
extern "C" void kernel_launch(void* const* d_in, const int* in_sizes, int n_in,
                              void* d_out, int out_size, void* d_ws, size_t ws_size,
                              hipStream_t stream) {
  (void)in_sizes; (void)n_in; (void)out_size; (void)ws_size;

  const float* x            = (const float*)d_in[0];
  const float* patch_w      = (const float*)d_in[1];
  const float* patch_b      = (const float*)d_in[2];
  const float* pe_norm_w    = (const float*)d_in[3];
  const float* pe_norm_b    = (const float*)d_in[4];
  const float* cls_token    = (const float*)d_in[5];
  const float* pos_embed    = (const float*)d_in[6];
  const float* norm1_w      = (const float*)d_in[7];
  const float* norm1_b      = (const float*)d_in[8];
  const float* qkv_w        = (const float*)d_in[9];
  const float* qkv_b        = (const float*)d_in[10];
  const float* proj_mat     = (const float*)d_in[11];
  const float* attn_proj_w  = (const float*)d_in[12];
  const float* attn_proj_b  = (const float*)d_in[13];
  const float* norm2_w      = (const float*)d_in[14];
  const float* norm2_b      = (const float*)d_in[15];
  const float* fc1_w        = (const float*)d_in[16];
  const float* fc1_b        = (const float*)d_in[17];
  const float* fc2_w        = (const float*)d_in[18];
  const float* fc2_b        = (const float*)d_in[19];
  const float* exit_w       = (const float*)d_in[20];
  const float* exit_b       = (const float*)d_in[21];
  const float* final_norm_w = (const float*)d_in[22];
  const float* final_norm_b = (const float*)d_in[23];
  const float* head_w       = (const float*)d_in[24];
  const float* head_b       = (const float*)d_in[25];
  float* out = (float*)d_out;

  // ---- workspace carving (256B aligned) ----
  char* ws = (char*)d_ws; size_t off = 0;
  auto take = [&](size_t bytes) -> void* {
    void* p = ws + off;
    off += (bytes + 255) & ~(size_t)255;
    return p;
  };
  u16*   wpeT   = (u16*)take((size_t)768 * 192 * 2);
  u16*   wqkv   = (u16*)take((size_t)12 * 192 * 576 * 2);
  u16*   wproj  = (u16*)take((size_t)12 * 256 * 64 * 2);
  u16*   wattnp = (u16*)take((size_t)12 * 192 * 192 * 2);
  u16*   wfc1   = (u16*)take((size_t)12 * 192 * 768 * 2);
  u16*   wfc2   = (u16*)take((size_t)12 * 768 * 192 * 2);
  u16*   wexit  = (u16*)take((size_t)3 * 192 * 1000 * 2);
  u16*   whead  = (u16*)take((size_t)192 * 1000 * 2);
  u16*   Ape    = (u16*)take((size_t)12544 * 768 * 2);
  float* tok    = (float*)take((size_t)12544 * 192 * 4);
  float* h      = (float*)take((size_t)12608 * 192 * 4);
  u16*   xn     = (u16*)take((size_t)12608 * 192 * 2);
  u16*   qkvb   = (u16*)take((size_t)12608 * 576 * 2);
  float* qn     = (float*)take((size_t)37824 * 4);
  float* kn     = (float*)take((size_t)37824 * 4);
  u16*   pqb    = (u16*)take((size_t)37824 * 256 * 2);
  u16*   pkb    = (u16*)take((size_t)37824 * 256 * 2);
  float* kc     = (float*)take((size_t)49152 * 4);
  float* di     = (float*)take((size_t)37824 * 4);
  u16*   attnb  = (u16*)take((size_t)192 * 197 * 224 * 2);  // row stride padded to 224
  u16*   outb   = (u16*)take((size_t)12608 * 192 * 2);
  u16*   mlp1   = (u16*)take((size_t)12608 * 768 * 2);
  u16*   pooled = (u16*)take((size_t)64 * 192 * 2);

  auto cvt = [&](const float* src, u16* dst, long long n) {
    k_f32_to_bf16<<<dim3((unsigned)((n + 255) / 256)), dim3(256), 0, stream>>>(src, dst, n);
  };

  // ---- weight conversion (f32 -> bf16) ----
  k_patchw_T<<<(768 * 192 + 255) / 256, 256, 0, stream>>>(patch_w, wpeT);
  cvt(qkv_w,       wqkv,   12LL * 192 * 576);
  cvt(proj_mat,    wproj,  12LL * 256 * 64);
  cvt(attn_proj_w, wattnp, 12LL * 192 * 192);
  cvt(fc1_w,       wfc1,   12LL * 192 * 768);
  cvt(fc2_w,       wfc2,   12LL * 768 * 192);
  cvt(exit_w,      wexit,  3LL * 192 * 1000);
  cvt(head_w,      whead,  192LL * 1000);

  // ---- patch embed: im2col + GEMM [12544,768]x[768,192] ----
  k_im2col<<<(unsigned)(((long long)12544 * 768 + 255) / 256), 256, 0, stream>>>(x, Ape);
  launch_gemm(stream, false, EPI_NONE,
              Ape, 0, 0, 768, wpeT, 0, 0, 192,
              tok, 0, 0, 192, nullptr, 0, 0, 0,
              patch_b, nullptr, 0, 0, 12544, 192, 768, 1, 1, 0.f);
  k_assemble<<<(12608 + 7) / 8, 256, 0, stream>>>(tok, cls_token, pos_embed,
                                                  pe_norm_w, pe_norm_b, h);

  const float SC = 0.35355339059327373f;  // 64^-0.25
  for (int i = 0; i < 12; ++i) {
    // LN1 -> xn (bf16)
    k_ln<<<1576, 256, 0, stream>>>(h, norm1_w + i * 192, norm1_b + i * 192, xn, 12608);
    // qkv GEMM [12608,192]x[192,576] (+bias) -> bf16
    launch_gemm(stream, false, EPI_NONE,
                xn, 0, 0, 192, wqkv + (size_t)i * 110592, 0, 0, 576,
                nullptr, 0, 0, 0, qkvb, 0, 0, 576,
                qkv_b + i * 576, nullptr, 0, 0, 12608, 576, 192, 1, 1, 0.f);
    k_rownorm<<<(37824 + 255) / 256, 256, 0, stream>>>(qkvb, qn, kn);
    // Performer feature maps: exp(q@pm^T - |q|^2/2)*d^-1/4  (batched over b,h)
    launch_gemm(stream, true, EPI_EXP,
                qkvb, 113472, 64, 576, wproj + (size_t)i * 16384, 0, 0, 64,
                nullptr, 0, 0, 0, pqb, 151296, 50432, 256,
                nullptr, qn, 591, 197, 197, 256, 64, 64, 3, SC);
    launch_gemm(stream, true, EPI_EXP,
                qkvb + 192, 113472, 64, 576, wproj + (size_t)i * 16384, 0, 0, 64,
                nullptr, 0, 0, 0, pkb, 151296, 50432, 256,
                nullptr, kn, 591, 197, 197, 256, 64, 64, 3, SC);
    k_kcsum<<<(49152 + 255) / 256, 256, 0, stream>>>(pkb, kc);
    k_dinv<<<(37824 + 255) / 256, 256, 0, stream>>>(pqb, kc, di);
    // attn = pq @ pk^T  [197,256]x[256,197] per (b,h) -> bf16 (ld 224)
    launch_gemm(stream, true, EPI_NONE,
                pqb, 151296, 50432, 256, pkb, 151296, 50432, 256,
                nullptr, 0, 0, 0, attnb, 132384, 44128, 224,
                nullptr, nullptr, 0, 0, 197, 197, 256, 64, 3, 0.f);
    // out = (attn @ v) * D_inv  (v strided inside qkv buffer) -> bf16 [12608,192]
    launch_gemm(stream, false, EPI_ROWSCALE,
                attnb, 132384, 44128, 224, qkvb + 384, 113472, 64, 576,
                nullptr, 0, 0, 0, outb, 37824, 64, 192,
                nullptr, di, 591, 197, 197, 64, 197, 64, 3, 0.f);
    // attention projection, residual-accumulated into h (fp32)
    launch_gemm(stream, false, EPI_ACCUM,
                outb, 0, 0, 192, wattnp + (size_t)i * 36864, 0, 0, 192,
                h, 0, 0, 192, nullptr, 0, 0, 0,
                attn_proj_b + i * 192, nullptr, 0, 0, 12608, 192, 192, 1, 1, 0.f);
    // MLP: LN2 -> fc1+GELU -> fc2 (+residual)
    k_ln<<<1576, 256, 0, stream>>>(h, norm2_w + i * 192, norm2_b + i * 192, xn, 12608);
    launch_gemm(stream, false, EPI_GELU,
                xn, 0, 0, 192, wfc1 + (size_t)i * 147456, 0, 0, 768,
                nullptr, 0, 0, 0, mlp1, 0, 0, 768,
                fc1_b + i * 768, nullptr, 0, 0, 12608, 768, 192, 1, 1, 0.f);
    launch_gemm(stream, false, EPI_ACCUM,
                mlp1, 0, 0, 768, wfc2 + (size_t)i * 147456, 0, 0, 192,
                h, 0, 0, 192, nullptr, 0, 0, 0,
                fc2_b + i * 192, nullptr, 0, 0, 12608, 192, 768, 1, 1, 0.f);
    // early exits after layers 4/8/12
    int e = (i == 3) ? 0 : (i == 7) ? 1 : (i == 11) ? 2 : -1;
    if (e >= 0) {
      k_pool<<<64, 192, 0, stream>>>(h, pooled);
      launch_gemm(stream, false, EPI_NONE,
                  pooled, 0, 0, 192, wexit + (size_t)e * 192000, 0, 0, 1000,
                  out + (size_t)(1 + e) * 64000, 0, 0, 1000, nullptr, 0, 0, 0,
                  exit_b + e * 1000, nullptr, 0, 0, 64, 1000, 192, 1, 1, 0.f);
    }
  }

  // final LN, then head on cls tokens (lda = 197*192 strides straight to cls rows)
  k_ln<<<1576, 256, 0, stream>>>(h, final_norm_w, final_norm_b, xn, 12608);
  launch_gemm(stream, false, EPI_NONE,
              xn, 0, 0, 37824, whead, 0, 0, 1000,
              out, 0, 0, 1000, nullptr, 0, 0, 0,
              head_b, nullptr, 0, 0, 64, 1000, 192, 1, 1, 0.f);
}
